// BiVRWKV_SpatialMix_CUDA_64639257805137
// MI455X (gfx1250) — matrix operand order repaired and checked by reading, not compiled, not run on hardware
//
#include <hip/hip_runtime.h>
#include <hip/hip_bf16.h>

// ---------------- problem constants ----------------
#define BDIM 16
#define CDIM 192
#define TDIM 3136              // 56*56
#define BT   (BDIM * TDIM)     // 50176
#define NCH  16                // chunks per sequence for the parallel scan
#define CHL  (TDIM / NCH)      // 196
#define LN_EPS  1e-5f
#define NEG_INF -1e38f

typedef __bf16 bf16;
typedef bf16  bf16x16 __attribute__((ext_vector_type(16)));
typedef bf16  bf16x8  __attribute__((ext_vector_type(8)));
typedef float f32x8   __attribute__((ext_vector_type(8)));

// A-matrix 16x32 bf16 fragment (ISA 7.12.2): lane 0-15 hold M=lane, K in
// {0..7, 16..23}; lanes 16-31 hold M=lane-16, K in {8..15, 24..31}.
static __device__ inline bf16x16 frag_a(const bf16* p) {
  bf16x8 lo = *(const bf16x8*)(p);
  bf16x8 hi = *(const bf16x8*)(p + 16);
  return __builtin_shufflevector(lo, hi, 0,1,2,3,4,5,6,7,8,9,10,11,12,13,14,15);
}
// B-matrix KxN fragment: lane holds column N=lane&15, K in {0..7, 16..23}
// (lanes 0-15) or {8..15, 24..31} (lanes 16-31), matching the A fragment.
static __device__ inline bf16x16 frag_b(const bf16* p) {
  bf16x8 lo = *(const bf16x8*)(p);
  bf16x8 hi = *(const bf16x8*)(p + 16);
  return __builtin_shufflevector(lo, hi, 0,1,2,3,4,5,6,7,8,9,10,11,12,13,14,15);
}

#define WMMA_BF16(A, Bf, Cacc) \
  __builtin_amdgcn_wmma_f32_16x16x32_bf16(false, (A), false, (Bf), (short)0, (Cacc), false, false)

// One 16(M) x 64(N) output tile for a single weight matrix; only 4 live
// accumulators (32 VGPRs) -> no spills. A fragments re-read per weight pass
// hit the WGP cache; L2-level A traffic is set by grid.y alone.
static __device__ inline void gemm_tile_16x64(const bf16* __restrict__ ap,
                                              const bf16* __restrict__ wgt,
                                              float* __restrict__ dst,
                                              int m0, int n0, int lane, int lr, int hiB) {
  const f32x8 z = {0.f,0.f,0.f,0.f,0.f,0.f,0.f,0.f};
  f32x8 acc[4];
#pragma unroll
  for (int nt = 0; nt < 4; ++nt) acc[nt] = z;
#pragma unroll
  for (int kk = 0; kk < CDIM; kk += 32) {
    bf16x16 a = frag_a(ap + kk);
#pragma unroll
    for (int nt = 0; nt < 4; ++nt) {
      size_t brow = (size_t)(n0 + nt * 16 + lr) * CDIM + hiB + kk;
      acc[nt] = WMMA_BF16(a, frag_b(wgt + brow), acc[nt]);
    }
  }
#pragma unroll
  for (int nt = 0; nt < 4; ++nt) {
#pragma unroll
    for (int i = 0; i < 8; ++i) {   // C/D layout: row = i + 8*(lane>>4), col = lane&15
      int row = m0 + i + ((lane >> 4) << 3);
      dst[(size_t)row * CDIM + n0 + nt * 16 + lr] = acc[nt][i];
    }
  }
}

// ---------------- prep: fp32 -> bf16 weight convert ----------------
__global__ void cvt_bf16(const float* __restrict__ s, bf16* __restrict__ d, int n) {
  int i = blockIdx.x * blockDim.x + threadIdx.x;
  if (i < n) d[i] = (bf16)s[i];
}

// ---------------- prep: x [B,C,T] f32 -> xs [B*T,C] bf16 (LDS tile transpose) --
__global__ void transpose_cast(const float* __restrict__ x, bf16* __restrict__ xs) {
  __shared__ float tile[32][33];
  int t0 = blockIdx.x * 32, c0 = blockIdx.y * 32, b = blockIdx.z;
#pragma unroll
  for (int j = 0; j < 4; ++j) {
    int c = c0 + threadIdx.y + j * 8;
    tile[threadIdx.y + j * 8][threadIdx.x] =
        x[((size_t)b * CDIM + c) * TDIM + t0 + threadIdx.x];
  }
  __syncthreads();
#pragma unroll
  for (int j = 0; j < 4; ++j) {
    int t = t0 + threadIdx.y + j * 8;
    xs[((size_t)b * TDIM + t) * CDIM + c0 + threadIdx.x] =
        (bf16)tile[threadIdx.x][threadIdx.y + j * 8];
  }
}

// ---------------- GEMM1: k,v,r = xs * W^T (WMMA bf16, f32 accum) ----------------
// block = 128 threads = 4 waves; wave owns 16 rows x 64 cols, three sequential
// weight passes (k, v, r) to keep live accumulators at 4.
__global__ void gemm_kvr(const bf16* __restrict__ xs,
                         const bf16* __restrict__ wk, const bf16* __restrict__ wv,
                         const bf16* __restrict__ wr,
                         float* __restrict__ ko, float* __restrict__ vo,
                         float* __restrict__ ro) {
  int lane = threadIdx.x & 31, wav = threadIdx.x >> 5;
  int m0 = (blockIdx.x * 4 + wav) * 16;
  int n0 = blockIdx.y * 64;
  int lr = lane & 15;
  int hiA = (lane >> 4) << 3;   // 0 / 8  element offset for A halves
  int hiB = (lane >> 4) << 3;   // 0 / 8  element offset for B halves
  const bf16* ap = xs + (size_t)(m0 + lr) * CDIM + hiA;

  gemm_tile_16x64(ap, wk, ko, m0, n0, lane, lr, hiB);
  gemm_tile_16x64(ap, wv, vo, m0, n0, lane, lr, hiB);
  gemm_tile_16x64(ap, wr, ro, m0, n0, lane, lr, hiB);
}

// ---------------- scan phase 1: per-chunk local WKV state (both directions) ----
__global__ void scan_chunk(const float* __restrict__ kbuf, const float* __restrict__ vbuf,
                           const float* __restrict__ wdec,
                           float* __restrict__ csp, float* __restrict__ csq,
                           float* __restrict__ cso) {
  int id  = blockIdx.x * blockDim.x + threadIdx.x;   // 2*B*NCH*C threads
  int c   = id % CDIM;
  int ch  = (id / CDIM) % NCH;
  int b   = (id / (CDIM * NCH)) % BDIM;
  int dir = id / (CDIM * NCH * BDIM);
  float w = -__expf(wdec[c]);
  float p = 0.f, q = 0.f, o = NEG_INF;
  int t = (dir == 0) ? ch * CHL : ch * CHL + CHL - 1;
  int step = (dir == 0) ? 1 : -1;
  size_t base = (size_t)b * TDIM * CDIM + c;
  for (int i = 0; i < CHL; ++i, t += step) {
    size_t idx = base + (size_t)t * CDIM;
    __builtin_prefetch(kbuf + idx + 16 * step * CDIM, 0, 1);
    __builtin_prefetch(vbuf + idx + 16 * step * CDIM, 0, 1);
    float kt = kbuf[idx], vt = vbuf[idx];
    float no = fmaxf(w + o, kt);
    float A2 = __expf(w + o - no);
    float B2 = __expf(kt - no);
    p = A2 * p + B2 * vt; q = A2 * q + B2; o = no;
  }
  csp[id] = p; csq[id] = q; cso[id] = o;
}

// ---------------- scan phase 2: chunk-prefix combine (16 steps/thread) ---------
__global__ void scan_combine(const float* __restrict__ wdec,
                             float* __restrict__ csp, float* __restrict__ csq,
                             float* __restrict__ cso) {
  int id = blockIdx.x * blockDim.x + threadIdx.x;    // 2*B*C threads
  if (id >= 2 * BDIM * CDIM) return;
  int c   = id % CDIM;
  int b   = (id / CDIM) % BDIM;
  int dir = id / (CDIM * BDIM);
  float w  = -__expf(wdec[c]);
  float lw = (float)CHL * w;
  float p = 0.f, q = 0.f, o = NEG_INF;
  for (int i = 0; i < NCH; ++i) {
    int ch = (dir == 0) ? i : (NCH - 1 - i);
    size_t idx = (((size_t)dir * BDIM + b) * NCH + ch) * CDIM + c;
    float cp = csp[idx], cq = csq[idx], co = cso[idx];
    csp[idx] = p; csq[idx] = q; cso[idx] = o;
    float no = fmaxf(o + lw, co);
    float ea = __expf(o + lw - no);
    float eb = __expf(co - no);
    p = ea * p + eb * cp; q = ea * q + eb * cq; o = no;
  }
}

// ---------------- scan phase 3: replay fwd then bwd within chunk, fuse gate ----
__global__ void scan_replay(const float* __restrict__ kbuf, const float* __restrict__ vbuf,
                            const float* __restrict__ rbuf,
                            const float* __restrict__ wdec, const float* __restrict__ ubuf,
                            const float* __restrict__ csp, const float* __restrict__ csq,
                            const float* __restrict__ cso,
                            float* __restrict__ hbuf) {
  int id = blockIdx.x * blockDim.x + threadIdx.x;    // B*NCH*C threads
  int c  = id % CDIM;
  int ch = (id / CDIM) % NCH;
  int b  = id / (CDIM * NCH);
  float w = -__expf(wdec[c]);
  float u = ubuf[c];
  size_t base = (size_t)b * TDIM * CDIM + c;
  // forward replay: stash y_fwd in h
  size_t s0 = (((size_t)0 * BDIM + b) * NCH + ch) * CDIM + c;
  float p = csp[s0], q = csq[s0], o = cso[s0];
  int t = ch * CHL;
  for (int i = 0; i < CHL; ++i, ++t) {
    size_t idx = base + (size_t)t * CDIM;
    __builtin_prefetch(kbuf + idx + 16 * CDIM, 0, 1);
    float kt = kbuf[idx], vt = vbuf[idx];
    float no = fmaxf(o, u + kt);
    float A  = __expf(o - no), Bw = __expf(u + kt - no);
    hbuf[idx] = (A * p + Bw * vt) / (A * q + Bw);
    float no2 = fmaxf(w + o, kt);
    float A2 = __expf(w + o - no2), B2 = __expf(kt - no2);
    p = A2 * p + B2 * vt; q = A2 * q + B2; o = no2;
  }
  // backward replay: combine + sigmoid gate in place
  size_t s1 = (((size_t)1 * BDIM + b) * NCH + ch) * CDIM + c;
  p = csp[s1]; q = csq[s1]; o = cso[s1];
  t = ch * CHL + CHL - 1;
  for (int i = 0; i < CHL; ++i, --t) {
    size_t idx = base + (size_t)t * CDIM;
    float kt = kbuf[idx], vt = vbuf[idx];
    float no = fmaxf(o, u + kt);
    float A  = __expf(o - no), Bw = __expf(u + kt - no);
    float yb = (A * p + Bw * vt) / (A * q + Bw);
    float sig = 1.f / (1.f + __expf(-rbuf[idx]));
    hbuf[idx] = sig * 0.5f * (hbuf[idx] + yb);
    float no2 = fmaxf(w + o, kt);
    float A2 = __expf(w + o - no2), B2 = __expf(kt - no2);
    p = A2 * p + B2 * vt; q = A2 * q + B2; o = no2;
  }
}

// ---------------- LayerNorm over C (wave32 per row) + bf16 cast ----------------
__global__ void layernorm_cast(const float* __restrict__ h, const float* __restrict__ g,
                               const float* __restrict__ beta, bf16* __restrict__ hln) {
  int lane = threadIdx.x & 31;
  int row  = blockIdx.x * 8 + (threadIdx.x >> 5);
  const float* hr = h + (size_t)row * CDIM;
  float v[6]; float s = 0.f, ss = 0.f;
#pragma unroll
  for (int j = 0; j < 6; ++j) { v[j] = hr[lane + 32 * j]; s += v[j]; ss += v[j] * v[j]; }
#pragma unroll
  for (int off = 16; off; off >>= 1) { s += __shfl_xor(s, off, 32); ss += __shfl_xor(ss, off, 32); }
  float mu = s * (1.f / CDIM);
  float var = ss * (1.f / CDIM) - mu * mu;
  float rs = rsqrtf(var + LN_EPS);
  bf16* dr = hln + (size_t)row * CDIM;
#pragma unroll
  for (int j = 0; j < 6; ++j) {
    int c = lane + 32 * j;
    dr[c] = (bf16)((v[j] - mu) * rs * g[c] + beta[c]);
  }
}

// ---------------- GEMM2: out[b,d,t] = (h_ln * Wo^T), transposed b128 stores ----
// Per-lane accumulator elements are 8 consecutive t values -> two float4 stores.
__global__ void gemm_out(const bf16* __restrict__ hln, const bf16* __restrict__ wo,
                         float* __restrict__ outp) {
  int lane = threadIdx.x & 31, wav = threadIdx.x >> 5;
  int m0 = (blockIdx.x * 4 + wav) * 16;
  int n0 = blockIdx.y * 64;
  int lr = lane & 15;
  int hiA = (lane >> 4) << 3;
  int hiB = (lane >> 4) << 3;
  const bf16* ap = hln + (size_t)(m0 + lr) * CDIM + hiA;

  const f32x8 z = {0.f,0.f,0.f,0.f,0.f,0.f,0.f,0.f};
  f32x8 acc[4];
#pragma unroll
  for (int nt = 0; nt < 4; ++nt) acc[nt] = z;

#pragma unroll
  for (int kk = 0; kk < CDIM; kk += 32) {
    bf16x16 a = frag_a(ap + kk);
#pragma unroll
    for (int nt = 0; nt < 4; ++nt) {
      size_t brow = (size_t)(n0 + nt * 16 + lr) * CDIM + hiB + kk;
      acc[nt] = WMMA_BF16(a, frag_b(wo + brow), acc[nt]);
    }
  }

  int bidx = m0 / TDIM;          // tiles never straddle batches (T % 64 == 0)
  int tl   = (m0 - bidx * TDIM) + ((lane >> 4) << 3);
#pragma unroll
  for (int nt = 0; nt < 4; ++nt) {
    int col = n0 + nt * 16 + lr;
    float* dst = outp + ((size_t)bidx * CDIM + col) * TDIM + tl;
    float4 s0 = make_float4(acc[nt][0], acc[nt][1], acc[nt][2], acc[nt][3]);
    float4 s1 = make_float4(acc[nt][4], acc[nt][5], acc[nt][6], acc[nt][7]);
    *(float4*)(dst)     = s0;
    *(float4*)(dst + 4) = s1;
  }
}

// ---------------- host launcher ----------------
extern "C" void kernel_launch(void* const* d_in, const int* in_sizes, int n_in,
                              void* d_out, int out_size, void* d_ws, size_t ws_size,
                              hipStream_t stream) {
  const float* x    = (const float*)d_in[0];
  const float* Wk   = (const float*)d_in[1];
  const float* Wv   = (const float*)d_in[2];
  const float* Wr   = (const float*)d_in[3];
  const float* Wo   = (const float*)d_in[4];
  const float* ln_g = (const float*)d_in[5];
  const float* ln_b = (const float*)d_in[6];
  const float* wdec = (const float*)d_in[7];
  const float* u    = (const float*)d_in[8];
  float* outp = (float*)d_out;

  char* ws = (char*)d_ws;
  size_t off = 0;
  bf16* xs  = (bf16*)(ws + off); off += (size_t)BT * CDIM * 2;
  bf16* wkb = (bf16*)(ws + off); off += (size_t)CDIM * CDIM * 2;
  bf16* wvb = (bf16*)(ws + off); off += (size_t)CDIM * CDIM * 2;
  bf16* wrb = (bf16*)(ws + off); off += (size_t)CDIM * CDIM * 2;
  bf16* wob = (bf16*)(ws + off); off += (size_t)CDIM * CDIM * 2;
  float* kb = (float*)(ws + off); off += (size_t)BT * CDIM * 4;
  float* vb = (float*)(ws + off); off += (size_t)BT * CDIM * 4;
  float* rb = (float*)(ws + off); off += (size_t)BT * CDIM * 4;
  float* hb = (float*)(ws + off); off += (size_t)BT * CDIM * 4;
  bf16* hln = (bf16*)(ws + off);  off += (size_t)BT * CDIM * 2;
  size_t csn = (size_t)2 * BDIM * NCH * CDIM;   // per-array chunk-state count
  float* csp = (float*)(ws + off); off += csn * 4;
  float* csq = (float*)(ws + off); off += csn * 4;
  float* cso = (float*)(ws + off); off += csn * 4;

  const int wn = CDIM * CDIM;
  cvt_bf16<<<(wn + 255) / 256, 256, 0, stream>>>(Wk, wkb, wn);
  cvt_bf16<<<(wn + 255) / 256, 256, 0, stream>>>(Wv, wvb, wn);
  cvt_bf16<<<(wn + 255) / 256, 256, 0, stream>>>(Wr, wrb, wn);
  cvt_bf16<<<(wn + 255) / 256, 256, 0, stream>>>(Wo, wob, wn);

  transpose_cast<<<dim3(TDIM / 32, CDIM / 32, BDIM), dim3(32, 8), 0, stream>>>(x, xs);

  gemm_kvr<<<dim3(BT / 64, CDIM / 64), 128, 0, stream>>>(xs, wkb, wvb, wrb, kb, vb, rb);

  scan_chunk<<<(2 * BDIM * NCH * CDIM) / 256, 256, 0, stream>>>(kb, vb, wdec, csp, csq, cso);
  scan_combine<<<(2 * BDIM * CDIM + 255) / 256, 256, 0, stream>>>(wdec, csp, csq, cso);
  scan_replay<<<(BDIM * NCH * CDIM) / 256, 256, 0, stream>>>(kb, vb, rb, wdec, u,
                                                             csp, csq, cso, hb);

  layernorm_cast<<<BT / 8, 256, 0, stream>>>(hb, ln_g, ln_b, hln);

  gemm_out<<<dim3(BT / 64, CDIM / 64), 128, 0, stream>>>(hln, wob, outp);
}
